// WindowAttentionClipRecurrent_72971494359722
// MI455X (gfx1250) — compile-verified, matrix operand order checked by reading, not run on hardware
//
#include <hip/hip_runtime.h>
#include <hip/hip_bf16.h>

// ---------------- problem constants ----------------
#define BB   2
#define TT   2
#define HH   56
#define WW2  56
#define CC   256
#define NHD  8
#define HD   32
#define SSZ  4
#define WSP  4
#define HSP  56
#define NWW  14
#define NWIN 28          // B * nWh * nWw
#define TQ   448         // T * H_sp * W_sp
#define NK   896         // 2 * TQ
#define NQT  28          // TQ / 16
#define NCH  28          // NK / 32 key chunks
#define NKT  56          // NK / 16 key tiles
// 32^-0.5 * log2(e): scores are pre-scaled so softmax uses pure exp2 (v_exp_f32)
#define SCALE_LOG2 0.2551260477024783f
#define PLANE (BB*TT*HH*WW2*CC)        // 3,211,264

typedef _Float16 v16h __attribute__((ext_vector_type(16)));
typedef _Float16 v8h  __attribute__((ext_vector_type(8)));
typedef float    v8f  __attribute__((ext_vector_type(8)));

union V16 { v16h v; v8h h[2]; };

// qkv element offset: ((((s*B + b)*T + t)*H + h)*W + w)*C + c
__device__ __forceinline__ int qkv_off(int s, int b, int t, int h, int w, int c) {
    return ((((s * BB + b) * TT + t) * HH + h) * WW2 + w) * CC + c;
}

// ---------------- kernel 1: pooled k/v (224 -> 4 linear per (b,ww,t,c)) ----------------
__global__ __launch_bounds__(256) void pool_kernel(const float* __restrict__ qkv,
                                                   const float* __restrict__ pool_w,
                                                   const float* __restrict__ pool_b,
                                                   float* __restrict__ p_k,
                                                   float* __restrict__ p_v) {
    int idx = blockIdx.x * 256 + threadIdx.x;          // B*nWw*T*C = 14336
    if (idx >= BB * NWW * TT * CC) return;
    int c  = idx & 255;
    int t  = (idx >> 8) & 1;
    int x  = idx >> 9;            // 0..27
    int ww = x % NWW;
    int b  = x / NWW;

    float sk0 = pool_b[0], sk1 = pool_b[1], sk2 = pool_b[2], sk3 = pool_b[3];
    float sv0 = pool_b[0], sv1 = pool_b[1], sv2 = pool_b[2], sv3 = pool_b[3];
    for (int n = 0; n < HSP * WSP; ++n) {
        int hs = n >> 2, ws = n & 3;
        float kv = qkv[qkv_off(1, b, t, hs, ww * 4 + ws, c)];
        float vv = qkv[qkv_off(2, b, t, hs, ww * 4 + ws, c)];
        float w0 = pool_w[0 * 224 + n], w1 = pool_w[1 * 224 + n];
        float w2 = pool_w[2 * 224 + n], w3 = pool_w[3 * 224 + n];
        sk0 += kv * w0; sk1 += kv * w1; sk2 += kv * w2; sk3 += kv * w3;
        sv0 += vv * w0; sv1 += vv * w1; sv2 += vv * w2; sv3 += vv * w3;
    }
    int base = ((((b * NWW + ww) * TT + t) * CC) + c) * SSZ;
    p_k[base + 0] = sk0; p_k[base + 1] = sk1; p_k[base + 2] = sk2; p_k[base + 3] = sk3;
    p_v[base + 0] = sv0; p_v[base + 1] = sv1; p_v[base + 2] = sv2; p_v[base + 3] = sv3;
}

// ---------------- kernel 2: LePE depthwise 3x3 conv (window-local padding) ----------------
__global__ __launch_bounds__(256) void lepe_kernel(const float* __restrict__ qkv,
                                                   const float* __restrict__ wgt,
                                                   const float* __restrict__ bias,
                                                   float* __restrict__ lepe) {
    int idx = blockIdx.x * 256 + threadIdx.x;          // NW*TQ*C = 3,211,264
    int c  = idx & 255;
    int r  = idx >> 8;
    int qi = r % TQ;
    int w  = r / TQ;
    int b  = w / NWW, ww = w % NWW;
    int t  = qi / 224;
    int rr = qi % 224;
    int hs = rr >> 2, ws = rr & 3;

    float acc = bias[c];
    #pragma unroll
    for (int dy = 0; dy < 3; ++dy) {
        int h2 = hs + dy - 1;
        if (h2 < 0 || h2 >= HSP) continue;
        #pragma unroll
        for (int dx = 0; dx < 3; ++dx) {
            int w2 = ws + dx - 1;
            if (w2 < 0 || w2 >= WSP) continue;   // padding at WINDOW boundary
            acc += qkv[qkv_off(2, b, t, h2, ww * 4 + w2, c)] * wgt[c * 9 + dy * 3 + dx];
        }
    }
    lepe[idx] = acc;
}

// helper: pooled-key fetch for key token j in [448,896) of window (b,ww), channel c
__device__ __forceinline__ float pooled_fetch(const float* __restrict__ pp,
                                              int b, int ww, int j, int c) {
    int jp = j - TQ;
    int t  = jp / 224;
    int kh = (jp % 224) / 56;
    int kw = jp % 56;
    int src = ww + kw - 28;
    if (src < 0 || src >= NWW) return 0.0f;
    return pp[((((b * NWW + src) * TT + t) * CC) + c) * SSZ + kh];
}

// ---------------- kernel 3: pack Q (A-operand layout, f16, pre-scaled) ----------------
__global__ __launch_bounds__(256) void pack_q_kernel(const float* __restrict__ qkv,
                                                     _Float16* __restrict__ qpack) {
    int idx = blockIdx.x * 256 + threadIdx.x;          // NW*NH*28*64 = 401408
    int h8   = idx & 1;
    int lane = (idx >> 1) & 31;
    int t0   = idx >> 6;
    int qt   = t0 % NQT;
    int t1   = t0 / NQT;
    int n    = t1 & 7;
    int w    = t1 >> 3;
    int b = w / NWW, ww = w % NWW;

    int m  = lane & 15;
    int K0 = ((lane < 16) ? 0 : 8) + (h8 ? 16 : 0);    // d base for these 8 halves
    int qi = qt * 16 + m;
    int t  = qi / 224;
    int rr = qi % 224;
    int hs = rr >> 2, ws = rr & 3;
    const float* src = qkv + qkv_off(0, b, t, hs, ww * 4 + ws, n * HD + K0);

    v8h out;
    #pragma unroll
    for (int i = 0; i < 8; ++i) out[i] = (_Float16)(src[i] * SCALE_LOG2);
    _Float16* dst = qpack + (size_t)((w * NHD + n) * NQT + qt) * 512 + lane * 16 + h8 * 8;
    *(v8h*)dst = out;
}

// ---------------- kernel 4: pack K (B-operand layout; regular + pooled keys) -----------
__global__ __launch_bounds__(256) void pack_k_kernel(const float* __restrict__ qkv,
                                                     const float* __restrict__ p_k,
                                                     _Float16* __restrict__ kpack) {
    int idx = blockIdx.x * 256 + threadIdx.x;          // NW*NH*56*64 = 802816
    int h8   = idx & 1;
    int lane = (idx >> 1) & 31;
    int t0   = idx >> 6;
    int kt   = t0 % NKT;
    int t1   = t0 / NKT;
    int n    = t1 & 7;
    int w    = t1 >> 3;
    int b = w / NWW, ww = w % NWW;

    int j  = kt * 16 + (lane & 15);                    // key column
    int K0 = ((lane < 16) ? 0 : 8) + (h8 ? 16 : 0);    // d (K-dim) base

    v8h out;
    if (j < TQ) {
        int t  = j / 224;
        int rr = j % 224;
        int hs = rr >> 2, ws = rr & 3;
        const float* src = qkv + qkv_off(1, b, t, hs, ww * 4 + ws, n * HD + K0);
        #pragma unroll
        for (int i = 0; i < 8; ++i) out[i] = (_Float16)src[i];
    } else {
        #pragma unroll
        for (int i = 0; i < 8; ++i)
            out[i] = (_Float16)pooled_fetch(p_k, b, ww, j, n * HD + K0 + i);
    }
    _Float16* dst = kpack + (size_t)((w * NHD + n) * NKT + kt) * 512 + lane * 16 + h8 * 8;
    *(v8h*)dst = out;
}

// ---------------- kernel 5: pack V (B-operand layout; key-major K dim) -----------------
__global__ __launch_bounds__(256) void pack_v_kernel(const float* __restrict__ qkv,
                                                     const float* __restrict__ p_v,
                                                     _Float16* __restrict__ vpack) {
    int idx = blockIdx.x * 256 + threadIdx.x;          // NW*NH*28*2*64 = 802816
    int h8   = idx & 1;
    int lane = (idx >> 1) & 31;
    int t0   = idx >> 6;
    int dt   = t0 & 1;
    int t2   = t0 >> 1;
    int ch   = t2 % NCH;
    int t1   = t2 / NCH;
    int n    = t1 & 7;
    int w    = t1 >> 3;
    int b = w / NWW, ww = w % NWW;

    int d  = dt * 16 + (lane & 15);                    // value column (head dim)
    int c  = n * HD + d;
    int K0 = ((lane < 16) ? 0 : 8) + (h8 ? 16 : 0);    // key offset base within chunk

    v8h out;
    #pragma unroll
    for (int i = 0; i < 8; ++i) {
        int j = ch * 32 + K0 + i;                      // key token
        float v;
        if (j < TQ) {
            int t  = j / 224;
            int rr = j % 224;
            int hs = rr >> 2, ws = rr & 3;
            v = qkv[qkv_off(2, b, t, hs, ww * 4 + ws, c)];
        } else {
            v = pooled_fetch(p_v, b, ww, j, c);
        }
        out[i] = (_Float16)v;
    }
    _Float16* dst = vpack + (size_t)(((w * NHD + n) * NCH + ch) * 2 + dt) * 512
                    + lane * 16 + h8 * 8;
    *(v8h*)dst = out;
}

// ---- bare v_max_num_f32 (no canonicalize insertion; scores are never sNaN) ----
__device__ __forceinline__ float asm_max(float a, float b) {
    float d;
    asm("v_max_num_f32 %0, %1, %2" : "=v"(d) : "v"(a), "v"(b));
    return d;
}

// ---- fused DPP16 row_ror max: one v_max_num_f32_dpp per step (src0 rotated) ----
#define DPP_MAX_ROR(v, imm)                                                     \
    asm("v_max_num_f32_dpp %0, %0, %0 row_ror:" imm                             \
        " row_mask:0xf bank_mask:0xf bound_ctrl:1" : "+v"(v))

__device__ __forceinline__ float grp16_max(float v) {
    DPP_MAX_ROR(v, "8");
    DPP_MAX_ROR(v, "4");
    DPP_MAX_ROR(v, "2");
    DPP_MAX_ROR(v, "1");
    return v;
}

// ---------------- kernel 6: flash attention, one wave per (window, head, qtile) --------
__global__ __launch_bounds__(128) void attn_kernel(const _Float16* __restrict__ qpack,
                                                   const _Float16* __restrict__ kpack,
                                                   const _Float16* __restrict__ vpack,
                                                   const float* __restrict__ lepe,
                                                   float* __restrict__ out) {
    __shared__ _Float16 sP[4][16 * 32];                // 1KB per wave: P tile staging
    int lane = threadIdx.x & 31;
    int wid  = threadIdx.x >> 5;
    int gid  = blockIdx.x * 4 + wid;                   // 6272 waves total
    int qt   = gid % NQT;
    int t1   = gid / NQT;
    int n    = t1 & 7;
    int w    = t1 >> 3;
    int b = w / NWW, ww = w % NWW;

    int m    = lane & 15;
    int koff = (lane < 16) ? 0 : 8;
    int mhi  = (lane < 16) ? 0 : 8;

    // Q A-operand (stays in registers for the whole pass)
    const _Float16* qp = qpack + (size_t)((w * NHD + n) * NQT + qt) * 512 + lane * 16;
    V16 aq; aq.h[0] = *(const v8h*)qp; aq.h[1] = *(const v8h*)(qp + 8);

    // all-ones B operand: P(16x32) x 1(32x16) puts rowsum(m) in every column -> lsum slots
    V16 ones;
    #pragma unroll
    for (int i = 0; i < 16; ++i) ones.v[i] = (_Float16)1.0f;

    const _Float16* kbase = kpack + (size_t)(w * NHD + n) * NKT * 512;
    const _Float16* vbase = vpack + (size_t)(w * NHD + n) * NCH * 2 * 512;

    v8f c0 = {}, c1 = {};
    float mrow[8], lsum[8];
    #pragma unroll
    for (int r = 0; r < 8; ++r) { mrow[r] = -1.0e30f; lsum[r] = 0.0f; }

    for (int ch = 0; ch < NCH; ++ch) {
        // ---- S = Q * K^T for 32 keys (two 16-key B tiles) ----
        const _Float16* kp0 = kbase + (size_t)(2 * ch) * 512 + lane * 16;
        V16 bk0, bk1;
        bk0.h[0] = *(const v8h*)kp0;         bk0.h[1] = *(const v8h*)(kp0 + 8);
        bk1.h[0] = *(const v8h*)(kp0 + 512); bk1.h[1] = *(const v8h*)(kp0 + 520);
        v8f z = {};
        v8f s0 = __builtin_amdgcn_wmma_f32_16x16x32_f16(false, aq.v, false, bk0.v,
                                                        (short)0, z, false, false);
        v8f s1 = __builtin_amdgcn_wmma_f32_16x16x32_f16(false, aq.v, false, bk1.v,
                                                        (short)0, z, false, false);
        // ---- online softmax: row max (fused DPP), base-2 exponentials ----
        float p0[8], p1[8], alpha[8];
        #pragma unroll
        for (int r = 0; r < 8; ++r) {
            float mx   = grp16_max(asm_max(s0[r], s1[r]));
            float mnew = asm_max(mrow[r], mx);
            alpha[r] = exp2f(mrow[r] - mnew);
            mrow[r] = mnew;
            p0[r] = exp2f(s0[r] - mnew);
            p1[r] = exp2f(s1[r] - mnew);
        }
        // ---- transpose P: D layout -> A layout via per-wave LDS tile ----
        #pragma unroll
        for (int r = 0; r < 8; ++r) {
            int mm = r + mhi;
            sP[wid][mm * 32 + (lane & 15)]      = (_Float16)p0[r];
            sP[wid][mm * 32 + 16 + (lane & 15)] = (_Float16)p1[r];
        }
        // Per-wave LDS region + DS ops are in-order within a wave: a wave-local
        // dscnt drain (plus compiler barrier) is enough; no block-wide barrier.
        asm volatile("s_wait_dscnt 0x0" ::: "memory");
        V16 ap;
        const _Float16* pb = &sP[wid][m * 32 + koff];
        ap.h[0] = *(const v8h*)pb;
        ap.h[1] = *(const v8h*)(pb + 16);
        asm volatile("" ::: "memory");        // keep next-iter stores below these loads
        // ---- row sums of P via WMMA against all-ones B (replaces DPP sum tree) ----
        v8f ts = __builtin_amdgcn_wmma_f32_16x16x32_f16(false, ap.v, false, ones.v,
                                                        (short)0, z, false, false);
        #pragma unroll
        for (int r = 0; r < 8; ++r) {
            lsum[r] = lsum[r] * alpha[r] + ts[r];
            c0[r] *= alpha[r];
            c1[r] *= alpha[r];
        }
        // ---- O += P * V (two 16-wide d tiles) ----
        const _Float16* vp0 = vbase + (size_t)(ch * 2) * 512 + lane * 16;
        V16 bv0, bv1;
        bv0.h[0] = *(const v8h*)vp0;         bv0.h[1] = *(const v8h*)(vp0 + 8);
        bv1.h[0] = *(const v8h*)(vp0 + 512); bv1.h[1] = *(const v8h*)(vp0 + 520);
        c0 = __builtin_amdgcn_wmma_f32_16x16x32_f16(false, ap.v, false, bv0.v,
                                                    (short)0, c0, false, false);
        c1 = __builtin_amdgcn_wmma_f32_16x16x32_f16(false, ap.v, false, bv1.v,
                                                    (short)0, c1, false, false);
    }

    // ---- epilogue: normalize (v_rcp), add LePE, scatter to (B,T,H,W,C) ----
    #pragma unroll
    for (int r = 0; r < 8; ++r) {
        int mm = r + mhi;
        int qi = qt * 16 + mm;
        int t  = qi / 224;
        int rr = qi % 224;
        int hs = rr >> 2, ws = rr & 3;
        float inv = __builtin_amdgcn_rcpf(lsum[r]);
        int cg = n * HD + (lane & 15);
        size_t lbase = ((size_t)w * TQ + qi) * CC;
        size_t obase = ((((size_t)(b * TT + t)) * HH + hs) * WW2 + (ww * 4 + ws)) * CC;
        out[obase + cg]      = c0[r] * inv + lepe[lbase + cg];
        out[obase + cg + 16] = c1[r] * inv + lepe[lbase + cg + 16];
    }
}

// ---------------- host launcher ----------------
extern "C" void kernel_launch(void* const* d_in, const int* in_sizes, int n_in,
                              void* d_out, int out_size, void* d_ws, size_t ws_size,
                              hipStream_t stream) {
    const float* qkv    = (const float*)d_in[0];
    const float* get_vw = (const float*)d_in[1];
    const float* get_vb = (const float*)d_in[2];
    const float* pool_w = (const float*)d_in[3];
    const float* pool_b = (const float*)d_in[4];
    float* out = (float*)d_out;

    // workspace carve (all 16B aligned)
    char* ws = (char*)d_ws;
    const size_t POOL_ELEMS = (size_t)BB * NWW * TT * CC * SSZ;     // 57344
    const size_t LEPE_ELEMS = (size_t)NWIN * TQ * CC;               // 3,211,264
    const size_t QP_ELEMS   = (size_t)NWIN * NHD * NQT * 512;       // 3,211,264 halves
    const size_t KP_ELEMS   = (size_t)NWIN * NHD * NKT * 512;       // 6,422,528 halves

    float*    p_k   = (float*)ws;                    ws += POOL_ELEMS * 4;
    float*    p_v   = (float*)ws;                    ws += POOL_ELEMS * 4;
    float*    lepe  = (float*)ws;                    ws += LEPE_ELEMS * 4;
    _Float16* qpack = (_Float16*)ws;                 ws += QP_ELEMS * 2;
    _Float16* kpack = (_Float16*)ws;                 ws += KP_ELEMS * 2;
    _Float16* vpack = (_Float16*)ws;                 /* end ~45.5 MB */

    pool_kernel<<<(BB * NWW * TT * CC + 255) / 256, 256, 0, stream>>>(
        qkv, pool_w, pool_b, p_k, p_v);
    lepe_kernel<<<(int)(LEPE_ELEMS / 256), 256, 0, stream>>>(
        qkv, get_vw, get_vb, lepe);
    pack_q_kernel<<<(int)((size_t)NWIN * NHD * NQT * 64 / 256), 256, 0, stream>>>(
        qkv, qpack);
    pack_k_kernel<<<(int)((size_t)NWIN * NHD * NKT * 64 / 256), 256, 0, stream>>>(
        qkv, p_k, kpack);
    pack_v_kernel<<<(int)((size_t)NWIN * NHD * NCH * 2 * 64 / 256), 256, 0, stream>>>(
        qkv, p_v, vpack);
    attn_kernel<<<NWIN * NHD * NQT / 4, 128, 0, stream>>>(
        qpack, kpack, vpack, lepe, out);
}
